// MoEGate_52338471469336
// MI455X (gfx1250) — matrix operand order repaired and checked by reading, not compiled, not run on hardware
//
#include <hip/hip_runtime.h>
#include <hip/hip_bf16.h>

typedef __attribute__((ext_vector_type(16))) __bf16 v16bf;
typedef __attribute__((ext_vector_type(8)))  float  v8f;

#define T_TOK   16384
#define HDIM    2048
#define NEXP    64
#define TOPK    8
#define KSTEPS  (HDIM / 32)       // 64 k-steps of 32
#define ROWPAD  36                // 32 floats + pad: 16B-aligned rows, conflict-free banks
#define LOGPAD  66                // 64 logits + pad

// CDNA5 async global->LDS copy path (ASYNCcnt), guarded so compile never breaks.
#if defined(__gfx1250__) && \
    __has_builtin(__builtin_amdgcn_global_load_async_to_lds_b128) && \
    __has_builtin(__builtin_amdgcn_s_wait_asynccnt)
#define USE_ASYNC_LDS 1
typedef int v4i_vs __attribute__((vector_size(16)));
typedef __attribute__((address_space(1))) v4i_vs* g_v4i_p;   // global (as1) int4*
typedef __attribute__((address_space(3))) v4i_vs* l_v4i_p;   // LDS (as3) int4*
#else
#define USE_ASYNC_LDS 0
#endif

// Split fp32 into bf16 hi (exact, truncated mantissa) + bf16 lo (remainder).
__device__ __forceinline__ void split_bf16(float x, __bf16& hi, __bf16& lo) {
    float xh = __uint_as_float(__float_as_uint(x) & 0xffff0000u);
    hi = (__bf16)xh;          // exact: low mantissa bits are zero
    lo = (__bf16)(x - xh);    // next ~8 bits of mantissa
}

// ---------------------------------------------------------------------------
// Kernel 1: weight [64][2048] fp32 -> hi/lo bf16 planes, pre-swizzled into the
// WMMA 16x16x32 B-fragment lanes, contiguous k order matching the A loads:
//   fragment(ks, nt): lane l holds column e = nt*16 + (l&15),
//   K = ks*32 + (l<16 ? 0 : 16) + 2*j + h for dword j, half h.
// ---------------------------------------------------------------------------
__global__ void moe_prep_weight(const float* __restrict__ w,
                                __bf16* __restrict__ bhi,
                                __bf16* __restrict__ blo) {
    int idx = blockIdx.x * blockDim.x + threadIdx.x;   // [0, NEXP*HDIM)
    int e = idx >> 11;
    int k = idx & (HDIM - 1);
    float x = w[idx];
    __bf16 hi, lo;
    split_bf16(x, hi, lo);
    int ks = k >> 5, kr = k & 31, nt = e >> 4;
    int lane = ((kr >= 16) ? 16 : 0) + (e & 15);
    int j = (kr & 15) >> 1, h = kr & 1;
    int o = (((ks * 4 + nt) * 32 + lane) << 4) + (j << 1) + h;
    bhi[o] = hi;
    blo[o] = lo;
}

// ---------------------------------------------------------------------------
// Kernel 2: logits GEMM via split-bf16 WMMA + softmax + top-8
// 256 threads = 8 wave32; block covers 128 tokens; wave covers 16x64 tile.
// ---------------------------------------------------------------------------
__global__ void __launch_bounds__(256)
moe_gate_kernel(const float* __restrict__ hs,
                const __bf16* __restrict__ bhi,
                const __bf16* __restrict__ blo,
                float* __restrict__ out_idx,
                float* __restrict__ out_w) {
    __shared__ float lds[2 * 128 * ROWPAD];   // 36 KB double-buffered A tile

    const int tid  = threadIdx.x;
    const int lane = tid & 31;
    const int wv   = tid >> 5;                // wave 0..7
    const int t0   = blockIdx.x * 128;        // first token of this block

    // Each thread owns 4 float4 slots of the 128x32 fp32 tile.
    int rowq[4], c4q[4];
#pragma unroll
    for (int q = 0; q < 4; ++q) {
        int f = tid + q * 256;                // float4 index in [0,1024)
        rowq[q] = f >> 3;
        c4q[q]  = f & 7;
    }

    // ---- preload A tile for ks=0 ----
#if USE_ASYNC_LDS
#pragma unroll
    for (int q = 0; q < 4; ++q) {
        const float* gp = hs + (size_t)(t0 + rowq[q]) * HDIM + c4q[q] * 4;
        float* lp = &lds[rowq[q] * ROWPAD + c4q[q] * 4];
        __builtin_amdgcn_global_load_async_to_lds_b128(
            (g_v4i_p)(void*)gp, (l_v4i_p)(void*)lp, 0, 0);
    }
    __builtin_amdgcn_s_wait_asynccnt(0);
#else
#pragma unroll
    for (int q = 0; q < 4; ++q) {
        float4 v = *(const float4*)(hs + (size_t)(t0 + rowq[q]) * HDIM + c4q[q] * 4);
        *(float4*)(&lds[rowq[q] * ROWPAD + c4q[q] * 4]) = v;
    }
#endif
    __syncthreads();

    v8f acc[4];
#pragma unroll
    for (int nt = 0; nt < 4; ++nt)
#pragma unroll
        for (int g = 0; g < 8; ++g) acc[nt][g] = 0.0f;

    const int off0 = (lane < 16) ? 0 : 16;        // A-fragment K offset for this lane
    const int arow = wv * 16 + (lane & 15);       // token row within block tile

    for (int ks = 0; ks < KSTEPS; ++ks) {
        const int cur = ks & 1, nxt = cur ^ 1;
        const float* ldc = &lds[cur * 128 * ROWPAD];

        // ---- stage next A tile: issued first, retired at end-of-iteration ----
#if USE_ASYNC_LDS
        if (ks + 1 < KSTEPS) {
#pragma unroll
            for (int q = 0; q < 4; ++q) {
                const float* gp = hs + (size_t)(t0 + rowq[q]) * HDIM +
                                  (ks + 1) * 32 + c4q[q] * 4;
                float* lp = &lds[nxt * 128 * ROWPAD + rowq[q] * ROWPAD + c4q[q] * 4];
                __builtin_amdgcn_global_load_async_to_lds_b128(
                    (g_v4i_p)(void*)gp, (l_v4i_p)(void*)lp, 0, 0);
            }
        }
#else
        float4 r[4];
        if (ks + 1 < KSTEPS) {
#pragma unroll
            for (int q = 0; q < 4; ++q) {
                r[q] = *(const float4*)(hs + (size_t)(t0 + rowq[q]) * HDIM +
                                        (ks + 1) * 32 + c4q[q] * 4);
            }
        }
#endif
        if (ks + 2 < KSTEPS) {  // HBM->L2 prefetch two steps ahead
            __builtin_prefetch(hs + (size_t)(t0 + rowq[1]) * HDIM + (ks + 2) * 32, 0, 0);
        }

        // ---- A fragment: 16 fp32 per lane from LDS, split into hi/lo bf16 ----
        float af[16];
        const float* rp = &ldc[arow * ROWPAD];
        *(float4*)&af[0]  = *(const float4*)(rp + off0);
        *(float4*)&af[4]  = *(const float4*)(rp + off0 + 4);
        *(float4*)&af[8]  = *(const float4*)(rp + off0 + 8);
        *(float4*)&af[12] = *(const float4*)(rp + off0 + 12);
        v16bf Ah, Al;
#pragma unroll
        for (int i = 0; i < 16; ++i) {
            __bf16 h, l;
            split_bf16(af[i], h, l);
            Ah[i] = h;
            Al[i] = l;
        }

        // ---- B fragments loaded per-nt (16 live VGPRs, no spill) + 3 WMMAs ----
#pragma unroll
        for (int nt = 0; nt < 4; ++nt) {
            int fo = (((ks * 4 + nt) * 32 + lane) << 4);
            v16bf Bh = *(const v16bf*)(bhi + fo);
            v16bf Bl = *(const v16bf*)(blo + fo);
            acc[nt] = __builtin_amdgcn_wmma_f32_16x16x32_bf16(
                false, Ah, false, Bh, (short)0, acc[nt], false, false);
            acc[nt] = __builtin_amdgcn_wmma_f32_16x16x32_bf16(
                false, Ah, false, Bl, (short)0, acc[nt], false, false);
            acc[nt] = __builtin_amdgcn_wmma_f32_16x16x32_bf16(
                false, Al, false, Bh, (short)0, acc[nt], false, false);
        }

        // ---- retire staging for next iteration ----
#if USE_ASYNC_LDS
        __builtin_amdgcn_s_wait_asynccnt(0);
#else
        if (ks + 1 < KSTEPS) {
            float* ldn = &lds[nxt * 128 * ROWPAD];
#pragma unroll
            for (int q = 0; q < 4; ++q)
                *(float4*)(&ldn[rowq[q] * ROWPAD + c4q[q] * 4]) = r[q];
        }
#endif
        __syncthreads();
    }

    // ---- epilogue: dump logits to LDS (reuse buffer), softmax + top-8 ----
    // C/D layout: VGPR g -> M = g + (lane>=16 ? 8 : 0), N = lane&15 within tile
    float* lg = lds;   // [128][LOGPAD] = 33 KB <= 36 KB
#pragma unroll
    for (int nt = 0; nt < 4; ++nt)
#pragma unroll
        for (int g = 0; g < 8; ++g) {
            int row = wv * 16 + g + ((lane >= 16) ? 8 : 0);
            int col = nt * 16 + (lane & 15);
            lg[row * LOGPAD + col] = acc[nt][g];
        }
    __syncthreads();

    if (tid < 128) {
        float* rowp = &lg[tid * LOGPAD];
        float mx = rowp[0];
#pragma unroll
        for (int e = 1; e < NEXP; ++e) mx = fmaxf(mx, rowp[e]);
        float denom = 0.0f;
#pragma unroll
        for (int e = 0; e < NEXP; ++e) denom += __expf(rowp[e] - mx);
        float inv = 1.0f / denom;

        size_t tok = (size_t)(t0 + tid);
#pragma unroll 1
        for (int kk = 0; kk < TOPK; ++kk) {
            float best = -3.4e38f;
            int bi = 0;
            for (int e = 0; e < NEXP; ++e) {
                float v = rowp[e];
                if (v > best) { best = v; bi = e; }   // strict > => lowest index on tie
            }
            rowp[bi] = -3.4e38f;
            out_idx[tok * TOPK + kk] = (float)bi;
            out_w[tok * TOPK + kk]   = __expf(best - mx) * inv;
        }
    }
}

extern "C" void kernel_launch(void* const* d_in, const int* in_sizes, int n_in,
                              void* d_out, int out_size, void* d_ws, size_t ws_size,
                              hipStream_t stream) {
    const float* hs = (const float*)d_in[0];   // [4,4096,2048] fp32
    const float* w  = (const float*)d_in[1];   // [64,2048] fp32
    // d_in[2] = top_k (fixed 8 in reference)

    __bf16* bhi = (__bf16*)d_ws;               // 256 KB
    __bf16* blo = bhi + (size_t)NEXP * HDIM;   // 256 KB
    float* out_idx = (float*)d_out;                       // [T,8] indices
    float* out_w   = out_idx + (size_t)T_TOK * TOPK;      // [T,8] weights

    moe_prep_weight<<<(NEXP * HDIM) / 256, 256, 0, stream>>>(w, bhi, blo);
    moe_gate_kernel<<<T_TOK / 128, 256, 0, stream>>>(hs, bhi, blo, out_idx, out_w);
}
